// QuantizedLinear_42949672986
// MI455X (gfx1250) — 8-bit matrix operand order repaired and checked by reading, compiled, not run on hardware
//
#include <hip/hip_runtime.h>
#include <stdint.h>

typedef __attribute__((ext_vector_type(8))) int v8i;

__global__ void pack_weights_kernel(const int* __restrict__ w,
                                    int8_t* __restrict__ w8,
                                    float* __restrict__ wsum,
                                    int IN) {
    const int o = blockIdx.x;
    const int t = threadIdx.x;
    __shared__ int red[256];

    int s = 0;
    for (int base = t * 4; base < IN; base += 256 * 4) {
        const int* p = w + (size_t)o * IN + base;
        int w0 = p[0], w1 = p[1], w2 = p[2], w3 = p[3];
        s += w0 + w1 + w2 + w3;
        int packed = (w0 & 0xFF) | ((w1 & 0xFF) << 8) |
                     ((w2 & 0xFF) << 16) | ((w3 & 0xFF) << 24);
        ((int*)w8)[((size_t)o * IN + base) >> 2] = packed;
    }
    red[t] = s;
    __syncthreads();
    for (int off = 128; off > 0; off >>= 1) {
        if (t < off) red[t] += red[t + off];
        __syncthreads();
    }
    if (t == 0) wsum[o] = (float)red[0];
}

__global__ void quantize_kernel(const float* __restrict__ x,
                                int8_t* __restrict__ x8,
                                const float* __restrict__ ascale,
                                const float* __restrict__ azp,
                                int total) {
    const int idx = blockIdx.x * blockDim.x + threadIdx.x;
    if (idx * 4 >= total) return;
    const float inv = 1.0f / ascale[0];
    const float zp  = azp[0];
    float4 v = ((const float4*)x)[idx];
    int q0 = (int)fminf(fmaxf(rintf(v.x * inv + zp), -128.0f), 127.0f);
    int q1 = (int)fminf(fmaxf(rintf(v.y * inv + zp), -128.0f), 127.0f);
    int q2 = (int)fminf(fmaxf(rintf(v.z * inv + zp), -128.0f), 127.0f);
    int q3 = (int)fminf(fmaxf(rintf(v.w * inv + zp), -128.0f), 127.0f);
    ((int*)x8)[idx] = (q0 & 0xFF) | ((q1 & 0xFF) << 8) |
                      ((q2 & 0xFF) << 16) | ((q3 & 0xFF) << 24);
}

__device__ __forceinline__ v8i load_a_frag(const int8_t* p  ) {
    v8i r;
#pragma unroll
    for (int j = 0; j < 2; ++j) {
        int4 t = *(const int4*)(p + 32 * j);
        r[4 * j]     = t.x;
        r[4 * j + 1] = t.y;
        r[4 * j + 2] = t.z;
        r[4 * j + 3] = t.w;
    }
    return r;
}

__device__ __forceinline__ v8i load_b_frag(const int8_t* p  ) {
    int4 t0 = *(const int4*)(p);
    int4 t1 = *(const int4*)(p + 32);
    v8i r;
    r[0] = t0.x; r[1] = t0.y; r[2] = t0.z; r[3] = t0.w;
    r[4] = t1.x; r[5] = t1.y; r[6] = t1.z; r[7] = t1.w;
    return r;
}

__global__ __launch_bounds__(256) void qgemm_wmma_kernel(
        const int8_t* __restrict__ x8,
        const int8_t* __restrict__ w8,
        const float* __restrict__ wsum,
        const float* __restrict__ wscale,
        const float* __restrict__ ascale,
        const float* __restrict__ azp,
        const float* __restrict__ bias,
        float* __restrict__ out,
        int M, int N, int K) {
    const int lane = threadIdx.x & 31;
    const int wid  = threadIdx.x >> 5;
    const int lm   = lane & 15;
    const int half = lane >> 4;
    const int wm   = wid & 3;
    const int wn   = wid >> 2;

    const int m_base = blockIdx.y * 128 + wm * 32;
    const int n_base = blockIdx.x * 128 + wn * 64;

    const int8_t* a0p = x8 + (size_t)(m_base + lm) * K + 16 * half;
    const int8_t* a1p = a0p + (size_t)16 * K;
    const int8_t* bp0 = w8 + (size_t)(n_base + 0 * 16 + lm) * K + 16 * half;
    const int8_t* bp1 = w8 + (size_t)(n_base + 1 * 16 + lm) * K + 16 * half;
    const int8_t* bp2 = w8 + (size_t)(n_base + 2 * 16 + lm) * K + 16 * half;
    const int8_t* bp3 = w8 + (size_t)(n_base + 3 * 16 + lm) * K + 16 * half;

    v8i acc[2][4] = {};

    for (int kk = 0; kk < K; kk += 64) {
        v8i a0 = load_a_frag(a0p + kk);
        v8i a1 = load_a_frag(a1p + kk);
        v8i b0 = load_b_frag(bp0 + kk);
        v8i b1 = load_b_frag(bp1 + kk);
        v8i b2 = load_b_frag(bp2 + kk);
        v8i b3 = load_b_frag(bp3 + kk);

        acc[0][0] = __builtin_amdgcn_wmma_i32_16x16x64_iu8(true, a0, true, b0, acc[0][0], false, false);
        acc[0][1] = __builtin_amdgcn_wmma_i32_16x16x64_iu8(true, a0, true, b1, acc[0][1], false, false);
        acc[0][2] = __builtin_amdgcn_wmma_i32_16x16x64_iu8(true, a0, true, b2, acc[0][2], false, false);
        acc[0][3] = __builtin_amdgcn_wmma_i32_16x16x64_iu8(true, a0, true, b3, acc[0][3], false, false);
        acc[1][0] = __builtin_amdgcn_wmma_i32_16x16x64_iu8(true, a1, true, b0, acc[1][0], false, false);
        acc[1][1] = __builtin_amdgcn_wmma_i32_16x16x64_iu8(true, a1, true, b1, acc[1][1], false, false);
        acc[1][2] = __builtin_amdgcn_wmma_i32_16x16x64_iu8(true, a1, true, b2, acc[1][2], false, false);
        acc[1][3] = __builtin_amdgcn_wmma_i32_16x16x64_iu8(true, a1, true, b3, acc[1][3], false, false);
    }

    const float sa = ascale[0];
    const float zp = azp[0];
#pragma unroll
    for (int ni = 0; ni < 4; ++ni) {
        const int n = n_base + 16 * ni + lm;
        const float sc  = wscale[n] * sa;
        const float off = zp * wsum[n];
        const float bi  = bias[n];
#pragma unroll
        for (int mi = 0; mi < 2; ++mi) {
#pragma unroll
            for (int r = 0; r < 8; ++r) {
                const int row = m_base + 16 * mi + 8 * half + r;
                out[(size_t)row * N + n] = ((float)acc[mi][ni][r] - off) * sc + bi;
            }
        }
    }
}

extern "C" void kernel_launch(void* const* d_in, const int* in_sizes, int n_in,
                              void* d_out, int out_size, void* d_ws, size_t ws_size,
                              hipStream_t stream) {
    const float* x      = (const float*)d_in[0];
    const int*   w      = (const int*)d_in[1];
    const float* wscale = (const float*)d_in[2];
    const float* ascale = (const float*)d_in[3];
    const float* azp    = (const float*)d_in[4];
    const float* bias   = (const float*)d_in[5];
    float* out = (float*)d_out;

    const int OUT = in_sizes[5];
    const int IN  = in_sizes[1] / OUT;
    const int M   = in_sizes[0] / IN;

    int8_t* x8 = (int8_t*)d_ws;
    int8_t* w8 = x8 + (size_t)M * IN;
    float*  wsum = (float*)(w8 + (size_t)OUT * IN);

    pack_weights_kernel<<<OUT, 256, 0, stream>>>(w, w8, wsum, IN);

    const int total = M * IN;
    quantize_kernel<<<(total / 4 + 255) / 256, 256, 0, stream>>>(x, x8, ascale, azp, total);

    dim3 grid(OUT / 128, M / 128);
    qgemm_wmma_kernel<<<grid, 256, 0, stream>>>(x8, w8, wsum, wscale, ascale, azp,
                                                bias, out, M, OUT, IN);
}
